// dMaSIFConv_14654428414389
// MI455X (gfx1250) — compile-verified, matrix operand order checked by reading, not run on hardware
//
#include <hip/hip_runtime.h>
#include <hip/hip_bf16.h>
#include <math.h>

typedef __attribute__((ext_vector_type(16))) _Float16 v16h;
typedef __attribute__((ext_vector_type(8)))  float    v8f;

#define NPTS   2048
#define HDIM   32
#define IDIM   16
#define NCUTS  8
#define NGROUPS 4
#define GN_EPS 1e-5f
// 1 / (sqrt(2) * 9.0)
#define PT_SCALE 0.07856742013183861f

// dynamic LDS: pn (NPTS*6 floats) + paired f_norm (NPTS*HDIM floats) = 304KB
#define PAIR_SMEM_FLOATS (NPTS * 6 + NPTS * HDIM)
#define PAIR_SMEM_BYTES  (PAIR_SMEM_FLOATS * 4)

// ---------------------------------------------------------------------------
// Small 2-layer MLP with leaky-relu (0.2 slope). One thread per point.
// ---------------------------------------------------------------------------
template <int IN_DIM>
__global__ void mlp2_kernel(const float* __restrict__ x,
                            const float* __restrict__ W1, const float* __restrict__ b1,
                            const float* __restrict__ W2, const float* __restrict__ b2,
                            float* __restrict__ y) {
  int n = blockIdx.x * blockDim.x + threadIdx.x;
  if (n >= NPTS) return;
  float xin[IN_DIM];
#pragma unroll
  for (int k = 0; k < IN_DIM; ++k) xin[k] = x[n * IN_DIM + k];
  float h[HDIM];
#pragma unroll 4
  for (int o = 0; o < HDIM; ++o) {
    float s = b1[o];
#pragma unroll
    for (int k = 0; k < IN_DIM; ++k) s = fmaf(W1[o * IN_DIM + k], xin[k], s);
    h[o] = (s >= 0.f) ? s : 0.2f * s;
  }
#pragma unroll 4
  for (int o = 0; o < HDIM; ++o) {
    float s = b2[o];
#pragma unroll
    for (int k = 0; k < HDIM; ++k) s = fmaf(W2[o * HDIM + k], h[k], s);
    y[n * HDIM + o] = (s >= 0.f) ? s : 0.2f * s;
  }
}

// ---------------------------------------------------------------------------
// GroupNorm statistics: one block per group; mean + biased var over 8ch x N.
// ---------------------------------------------------------------------------
__global__ void gn_stats_kernel(const float* __restrict__ x, float* __restrict__ stats) {
  const int g = blockIdx.x;
  const int tid = threadIdx.x;
  const int CH = HDIM / NGROUPS;            // 8
  const int M = CH * NPTS;                  // 16384
  float s = 0.f, ss = 0.f;
  for (int e = tid; e < M; e += blockDim.x) {
    int n = e >> 3;
    int c = g * CH + (e & 7);
    float v = x[n * HDIM + c];
    s += v;
    ss = fmaf(v, v, ss);
  }
  __shared__ float rs[256], rss[256];
  rs[tid] = s; rss[tid] = ss;
  __syncthreads();
  for (int off = 128; off > 0; off >>= 1) {
    if (tid < off) { rs[tid] += rs[tid + off]; rss[tid] += rss[tid + off]; }
    __syncthreads();
  }
  if (tid == 0) {
    float mu = rs[0] / (float)M;
    float var = rss[0] / (float)M - mu * mu;
    stats[g * 2 + 0] = mu;
    stats[g * 2 + 1] = rsqrtf(var + GN_EPS);
  }
}

// Standard-layout GroupNorm apply (used for the final output).
__global__ void gn_apply_kernel(const float* __restrict__ x, const float* __restrict__ stats,
                                const float* __restrict__ gamma, const float* __restrict__ beta,
                                float* __restrict__ y) {
  int idx = blockIdx.x * blockDim.x + threadIdx.x;
  if (idx >= NPTS * HDIM) return;
  int c = idx & (HDIM - 1);
  int g = c >> 3;
  float mu = stats[g * 2 + 0];
  float rs = stats[g * 2 + 1];
  y[idx] = fmaf((x[idx] - mu) * rs, gamma[c], beta[c]);
}

// GroupNorm apply writing the channel-paired layout consumed by the pair
// kernel: y[n*32 + c*2 + 0] = fn(n, c), y[n*32 + c*2 + 1] = fn(n, c+16).
__global__ void gn_apply_pair_kernel(const float* __restrict__ x, const float* __restrict__ stats,
                                     const float* __restrict__ gamma, const float* __restrict__ beta,
                                     float* __restrict__ y) {
  int idx = blockIdx.x * blockDim.x + threadIdx.x;
  if (idx >= NPTS * 16) return;
  int n = idx >> 4;
  int c = idx & 15;
  int c2 = c + 16;
  int g0 = c >> 3, g1 = c2 >> 3;
  float mu0 = stats[g0 * 2 + 0], rs0 = stats[g0 * 2 + 1];
  float mu1 = stats[g1 * 2 + 0], rs1 = stats[g1 * 2 + 1];
  float v0 = fmaf((x[n * HDIM + c]  - mu0) * rs0, gamma[c],  beta[c]);
  float v1 = fmaf((x[n * HDIM + c2] - mu1) * rs1, gamma[c2], beta[c2]);
  float2 out = make_float2(v0, v1);
  *(float2*)&y[n * HDIM + (c << 1)] = out;
}

// ---------------------------------------------------------------------------
// All-pairs quasi-geodesic convolution. One wave32 per output point i,
// 32 waves (1024 threads) per block sharing a 304KB LDS image of the point
// cloud (scaled xyz + normal) and the channel-paired normalized features.
// Tiles of 16 j-points form the M dimension of v_wmma_f32_16x16x32_f16
// (K = 8 real cuts + zero pad); B operands hold the constant A2 weights.
// D layout: lane holds channel n=lane&15, pair m = v + 8*(lane>>4).
// ---------------------------------------------------------------------------
__global__ void __launch_bounds__(1024)
pair_conv_kernel(const float* __restrict__ points,
                 const float* __restrict__ nuv,
                 const float* __restrict__ A1, const float* __restrict__ B1,
                 const float* __restrict__ A2, const float* __restrict__ B2,
                 const float* __restrict__ fpair,   // channel-paired f_norm
                 float* __restrict__ out) {
  extern __shared__ float smem[];
  float* pn = smem;                       // NPTS*6
  float* sf = smem + NPTS * 6;            // NPTS*HDIM (paired)

  const int tid = threadIdx.x;
  for (int p = tid; p < NPTS; p += blockDim.x) {
    pn[p * 6 + 0] = points[p * 3 + 0] * PT_SCALE;
    pn[p * 6 + 1] = points[p * 3 + 1] * PT_SCALE;
    pn[p * 6 + 2] = points[p * 3 + 2] * PT_SCALE;
    pn[p * 6 + 3] = nuv[p * 9 + 0];
    pn[p * 6 + 4] = nuv[p * 9 + 1];
    pn[p * 6 + 5] = nuv[p * 9 + 2];
  }
  {
    float4* dst = (float4*)sf;
    const float4* src = (const float4*)fpair;
    for (int k = tid; k < (NPTS * HDIM) / 4; k += blockDim.x) dst[k] = src[k];
  }
  __syncthreads();

  const int lane = tid & 31;
  const int wid  = tid >> 5;
  const int half = lane >> 4;
  const int col  = lane & 15;
  const int i = blockIdx.x * (blockDim.x >> 5) + wid;   // uniform per wave
  if (i >= NPTS) return;

  // per-wave (uniform) constants for point i
  const float pix = points[i * 3 + 0] * PT_SCALE;
  const float piy = points[i * 3 + 1] * PT_SCALE;
  const float piz = points[i * 3 + 2] * PT_SCALE;
  const float u0 = nuv[i * 9 + 0], u1 = nuv[i * 9 + 1], u2 = nuv[i * 9 + 2];
  const float u3 = nuv[i * 9 + 3], u4 = nuv[i * 9 + 4], u5 = nuv[i * 9 + 5];
  const float u6 = nuv[i * 9 + 6], u7 = nuv[i * 9 + 7], u8 = nuv[i * 9 + 8];

  float a1r[NCUTS][3], b1r[NCUTS];
#pragma unroll
  for (int c = 0; c < NCUTS; ++c) {
    a1r[c][0] = A1[c * 3 + 0];
    a1r[c][1] = A1[c * 3 + 1];
    a1r[c][2] = A1[c * 3 + 2];
    b1r[c] = B1[c];
  }

  // B operands: A2 weights for channel halves [0,16) and [16,32).
  v16h bm0 = {};
  v16h bm1 = {};
#pragma unroll
  for (int e = 0; e < 8; ++e) {
    float w0 = (half == 0) ? A2[col * NCUTS + e] : 0.f;          // K=e, N=col
    float w1 = (half == 0) ? A2[(col + 16) * NCUTS + e] : 0.f;
    bm0[e] = (_Float16)w0;
    bm1[e] = (_Float16)w1;
  }
  const float b20 = B2[col];
  const float b21 = B2[col + 16];
  const v8f zc = {};

  float acc0 = 0.f, acc1 = 0.f;

  for (int j0 = 0; j0 < NPTS; j0 += 16) {
    const int j = j0 + col;                 // both halves evaluate pair "col"
    const float* q = &pn[j * 6];
    const float dx = q[0] - pix, dy = q[1] - piy, dz = q[2] - piz;
    const float sq = fmaf(dx, dx, fmaf(dy, dy, dz * dz));
    const float dot = fmaf(u0, q[3], fmaf(u1, q[4], u2 * q[5]));
    const float tm = 2.f - dot;
    const float w = __expf(-sq * tm * tm);
    const float X0 = fmaf(u0, dx, fmaf(u1, dy, u2 * dz));
    const float X1 = fmaf(u3, dx, fmaf(u4, dy, u5 * dz));
    const float X2 = fmaf(u6, dx, fmaf(u7, dy, u8 * dz));

    v16h am = {};
#pragma unroll
    for (int c = 0; c < NCUTS; ++c) {
      float xc = fmaf(a1r[c][0], X0, fmaf(a1r[c][1], X1, fmaf(a1r[c][2], X2, b1r[c])));
      xc = fmaxf(xc, 0.f);
      am[c] = (_Float16)((half == 0) ? xc : 0.f);   // lanes>=16 carry K=8..15 pad
    }

    // D = A(16 pairs x K8) * A2^T  for both channel halves
    v8f d0 = __builtin_amdgcn_wmma_f32_16x16x32_f16(false, am, false, bm0,
                                                    (short)0, zc, false, false);
    v8f d1 = __builtin_amdgcn_wmma_f32_16x16x32_f16(false, am, false, bm1,
                                                    (short)0, zc, false, false);

    const float* frow = &sf[j0 * HDIM + (col << 1)];   // ds, immediate offsets
#pragma unroll
    for (int v = 0; v < 8; ++v) {
      const int m = v + 8 * half;                       // pair index of D[v]
      const float wv = __shfl(w, m, 32);                // ds_bpermute broadcast
      const float2 fp = *(const float2*)&frow[m * HDIM];
      acc0 = fmaf(wv * fp.x, fmaxf(d0[v] + b20, 0.f), acc0);
      acc1 = fmaf(wv * fp.y, fmaxf(d1[v] + b21, 0.f), acc1);
    }
  }

  // fold the two half-wave partial sums (pairs 0-7 vs 8-15)
  acc0 += __shfl_xor(acc0, 16, 32);
  acc1 += __shfl_xor(acc1, 16, 32);
  if (half == 0) {
    out[i * HDIM + col] = acc0;
    out[i * HDIM + col + 16] = acc1;
  }
}

// ---------------------------------------------------------------------------
extern "C" void kernel_launch(void* const* d_in, const int* in_sizes, int n_in,
                              void* d_out, int out_size, void* d_ws, size_t ws_size,
                              hipStream_t stream) {
  (void)in_sizes; (void)n_in; (void)out_size; (void)ws_size;
  const float* points  = (const float*)d_in[0];
  const float* nuv     = (const float*)d_in[1];
  const float* feats   = (const float*)d_in[2];
  const float* W_in1   = (const float*)d_in[3];
  const float* b_in1   = (const float*)d_in[4];
  const float* W_in2   = (const float*)d_in[5];
  const float* b_in2   = (const float*)d_in[6];
  const float* g_in_w  = (const float*)d_in[7];
  const float* g_in_b  = (const float*)d_in[8];
  const float* A1      = (const float*)d_in[9];
  const float* A2      = (const float*)d_in[10];
  const float* W_out1  = (const float*)d_in[11];
  const float* b_out1  = (const float*)d_in[12];
  const float* W_out2  = (const float*)d_in[13];
  const float* b_out2  = (const float*)d_in[14];
  const float* g_out_w = (const float*)d_in[15];
  const float* g_out_b = (const float*)d_in[16];
  const float* B1      = (const float*)d_in[17];
  const float* B2      = (const float*)d_in[18];

  float* bufA  = (float*)d_ws;              // NPTS*HDIM floats (f_pre, later conv sums)
  float* bufB  = bufA + NPTS * HDIM;        // NPTS*HDIM floats (paired f_norm, later g_pre)
  float* stats = bufB + NPTS * HDIM;        // 8 floats
  float* outF  = (float*)d_out;

  // 1) net_in + GroupNorm -> bufB (channel-paired f_norm)
  mlp2_kernel<IDIM><<<(NPTS + 255) / 256, 256, 0, stream>>>(feats, W_in1, b_in1,
                                                            W_in2, b_in2, bufA);
  gn_stats_kernel<<<NGROUPS, 256, 0, stream>>>(bufA, stats);
  gn_apply_pair_kernel<<<(NPTS * 16 + 255) / 256, 256, 0, stream>>>(bufA, stats,
                                                                    g_in_w, g_in_b, bufB);
  // 2) all-pairs WMMA convolution -> bufA (per-point sums, standard layout)
  pair_conv_kernel<<<NPTS / 32, 1024, PAIR_SMEM_BYTES, stream>>>(points, nuv, A1, B1,
                                                                 A2, B2, bufB, bufA);
  // 3) net_out + GroupNorm -> d_out
  mlp2_kernel<HDIM><<<(NPTS + 255) / 256, 256, 0, stream>>>(bufA, W_out1, b_out1,
                                                            W_out2, b_out2, bufB);
  gn_stats_kernel<<<NGROUPS, 256, 0, stream>>>(bufB, stats);
  gn_apply_kernel<<<(NPTS * HDIM + 255) / 256, 256, 0, stream>>>(bufB, stats,
                                                                 g_out_w, g_out_b, outF);
}
